// GroupQueryAttention_9981503996431
// MI455X (gfx1250) — matrix operand order repaired and checked by reading, not compiled, not run on hardware
//
#include <hip/hip_runtime.h>

typedef __attribute__((ext_vector_type(16))) __bf16 v16bf;
typedef __attribute__((ext_vector_type(8)))  __bf16 v8bf;
typedef __attribute__((ext_vector_type(8)))  float  v8f;

#define BQ   2
#define SQ   2048
#define EQ   1024
#define HQ   16
#define GQ   4
#define HDQ  64
#define GHQ  4
#define KVQ  256
#define MTOT (BQ * SQ)   // 4096 rows total
#define AW   4           // attention waves per block (share one K stream)

// ---------- fragment load helpers (contiguous k order, K = k0 + 16h + e, same as B) ----------

// A-matrix 16x32 bf16: lane holds row m=lane%16; elements 0..7 -> K=k0+16h+e,
// elements 8..15 -> K=k0+16h+8+e  (h = lane>=16). Two contiguous 8-elem chunks.
static __device__ inline v16bf frag_a_2x8(const __bf16* p0, const __bf16* p1) {
  union { v16bf v; v8bf h[2]; } u;
  u.h[0] = *(const v8bf*)p0;
  u.h[1] = *(const v8bf*)p1;
  return u.v;
}

// ---------- f32 -> bf16 converts ----------

__global__ void cvt_bf16_kernel(const float* __restrict__ in,
                                __bf16* __restrict__ out, int n) {
  int i = blockIdx.x * blockDim.x + threadIdx.x;
  if (i < n) out[i] = (__bf16)in[i];
}

// W is [K][N] row-major f32; write Wt as [N][K] bf16 (so WMMA B-frags are contiguous).
__global__ void cvt_wT_kernel(const float* __restrict__ W,
                              __bf16* __restrict__ Wt, int K, int N) {
  int i = blockIdx.x * blockDim.x + threadIdx.x;
  if (i >= K * N) return;
  int nIdx = i / K;
  int kIdx = i % K;
  Wt[i] = (__bf16)W[(size_t)kIdx * N + nIdx];
}

// ---------- GEMM: C[M,N] = A[M,K](bf16) * Wt[N,K](bf16)^T + bias ----------
// 8 waves/block, each wave computes a 16x64 tile; software-pipelined fragment
// loads so VMEM overlaps v_wmma_f32_16x16x32_bf16.
// mode 0: bf16 out, C[row*ldc+col]
// mode 1: bf16 out transposed, C[col*M+row]    (used to produce V^T)
// mode 2: f32 out, C[row*ldc+col]

__global__ void gemm_bf16_wmma(const __bf16* __restrict__ A,
                               const __bf16* __restrict__ Wt,
                               const float* __restrict__ bias,
                               void* __restrict__ outp,
                               int M, int N, int K, int ldc, int mode) {
  const int lane = threadIdx.x & 31;
  const int wave = threadIdx.x >> 5;
  const int h    = lane >> 4;       // half-wave select
  const int n16  = lane & 15;

  const int tilesN = N >> 6;                    // 64-wide tiles
  const int tile   = blockIdx.x * 8 + wave;
  if (tile >= (M >> 4) * tilesN) return;
  const int tm = tile / tilesN;
  const int tn = tile % tilesN;

  v8f acc[4] = {};
  const __bf16* arow = A + (size_t)(tm * 16 + n16) * K;
  const __bf16* brow[4];
#pragma unroll
  for (int j = 0; j < 4; ++j)
    brow[j] = Wt + (size_t)(tn * 64 + j * 16 + n16) * K + 16 * h;

  // ---- software pipeline: prime k=0, then load k+32 while computing k ----
  v16bf a_cur = frag_a_2x8(arow + 16 * h, arow + 16 * h + 8);
  v16bf b_cur[4];
#pragma unroll
  for (int j = 0; j < 4; ++j) b_cur[j] = *(const v16bf*)(brow[j]);

  for (int k0 = 0; k0 < K; k0 += 32) {
    const int kn = (k0 + 32 < K) ? (k0 + 32) : 0;
    v16bf a_nxt = frag_a_2x8(arow + kn + 16 * h, arow + kn + 16 * h + 8);
    v16bf b_nxt[4];
#pragma unroll
    for (int j = 0; j < 4; ++j) b_nxt[j] = *(const v16bf*)(brow[j] + kn);
    __builtin_prefetch(arow + k0 + 512, 0, 1);   // global_prefetch_b8

#pragma unroll
    for (int j = 0; j < 4; ++j) {
      acc[j] = __builtin_amdgcn_wmma_f32_16x16x32_bf16(
          false, a_cur, false, b_cur[j], (short)0, acc[j], false, false);
    }
    a_cur = a_nxt;
#pragma unroll
    for (int j = 0; j < 4; ++j) b_cur[j] = b_nxt[j];
  }

#pragma unroll
  for (int j = 0; j < 4; ++j) {
    const int col = tn * 64 + j * 16 + n16;
    const float bv = bias[col];
#pragma unroll
    for (int r = 0; r < 8; ++r) {
      const int row = tm * 16 + r + 8 * h;      // C layout: M = r + 8h
      const float val = acc[j][r] + bv;
      if (mode == 0) {
        ((__bf16*)outp)[(size_t)row * ldc + col] = (__bf16)val;
      } else if (mode == 1) {
        ((__bf16*)outp)[(size_t)col * M + row] = (__bf16)val;
      } else {
        ((float*)outp)[(size_t)row * ldc + col] = val;
      }
    }
  }
}

// ---------- flash-style GQA attention ----------
// 4 waves/block; ids packed so all 4 waves share (b,g,gh) and consecutive query
// tiles -> the 32x64 K tile is staged into LDS ONCE per block via
// GLOBAL_LOAD_ASYNC_TO_LDS_B128 (double-buffered, ASYNCcnt-tracked) and shared.
// Qb: [B*S, E] bf16   Kb: [B*S, KV] bf16   Vt: [G*HD, B*S] bf16   Ob: [B*S, E] bf16

__global__ void gqa_attn_wmma(const __bf16* __restrict__ Qb,
                              const __bf16* __restrict__ Kb,
                              const __bf16* __restrict__ Vt,
                              __bf16* __restrict__ Ob) {
  __shared__ __attribute__((aligned(32))) __bf16 kst[2][32 * HDQ];   // 2 x 4KB K stage
  __shared__ __attribute__((aligned(32))) __bf16 plds[AW][16 * 32];  // P re-layout

  const int lane = threadIdx.x & 31;
  const int wave = threadIdx.x >> 5;
  const int h    = lane >> 4;
  const int n16  = lane & 15;

  int id = blockIdx.x * AW + wave;
  const int ST = SQ / 16;           // 128 query tiles per (b,g,gh); AW divides 128
  const int sT = id % ST; id /= ST;
  const int gh = id % GHQ; id /= GHQ;
  const int g  = id % GQ;  id /= GQ;
  const int b  = id;                // 0..B-1
  const int head = g * GHQ + gh;

  // Q fragments for the whole 16x64 query tile (reused every key step)
  const __bf16* qrow = Qb + (size_t)(b * SQ + sT * 16 + n16) * EQ + head * HDQ;
  const v16bf qa0 = frag_a_2x8(qrow + 16 * h,      qrow + 16 * h + 8);
  const v16bf qa1 = frag_a_2x8(qrow + 32 + 16 * h, qrow + 40 + 16 * h);

  const __bf16* kbase = Kb + (size_t)b * SQ * KVQ + g * HDQ;
  const __bf16* vbase = Vt + (size_t)(g * HDQ) * MTOT + (size_t)b * SQ;

  // async-stage one 32x64 bf16 K tile: 4KB = 8 x (32 lanes x 16B)
  auto stageK = [&](int bufIdx, int t0) {
#pragma unroll
    for (int i = 0; i < 8; ++i) {
      const int c   = i * 32 + lane;          // 16B chunk id, 8 chunks per row
      const int row = c >> 3;
      const int cir = c & 7;
      const __bf16* gp = kbase + (size_t)(t0 + row) * KVQ + cir * 8;
      const unsigned loff = (unsigned)(size_t)&kst[bufIdx][row * HDQ + cir * 8];
      asm volatile("global_load_async_to_lds_b128 %0, %1, off"
                   :: "v"(loff), "v"(gp) : "memory");
    }
  };

  v8f o[4] = {};
  float m[8], l[8];
#pragma unroll
  for (int r = 0; r < 8; ++r) { m[r] = -1e30f; l[r] = 0.0f; }
  const float scale = 0.125f;       // 1/sqrt(HD)

  if (wave == 0) stageK(0, 0);      // prime the pipeline
  int buf = 0;

  for (int t0 = 0; t0 < SQ; t0 += 32) {
    __syncthreads();                // all waves done reading buf^1 from last iter
    if (wave == 0) {
      if (t0 + 32 < SQ) {
        stageK(buf ^ 1, t0 + 32);   // prefetch next tile while this one computes
        asm volatile("s_wait_asynccnt 0x8" ::: "memory");  // current tile landed
      } else {
        asm volatile("s_wait_asynccnt 0x0" ::: "memory");
      }
    }
    __syncthreads();                // K tile visible to all waves

    // ---- V fragments issued early so they fly during QK^T + softmax
    const __bf16* vr = vbase + t0 + 16 * h;
    v16bf vf[4];
#pragma unroll
    for (int j = 0; j < 4; ++j)
      vf[j] = *(const v16bf*)(vr + (size_t)(j * 16 + n16) * MTOT);

    // ---- scores from the shared LDS K tile: two 16x16 tiles, K-dim 64 -> 2 WMMA each
    v8f s0 = {}, s1 = {};
    {
      const __bf16* kr0 = &kst[buf][n16 * HDQ + 16 * h];
      v16bf b0 = *(const v16bf*)kr0;          // d = 0..31
      v16bf b1 = *(const v16bf*)(kr0 + 32);   // d = 32..63
      s0 = __builtin_amdgcn_wmma_f32_16x16x32_bf16(false, qa0, false, b0, (short)0, s0, false, false);
      s0 = __builtin_amdgcn_wmma_f32_16x16x32_bf16(false, qa1, false, b1, (short)0, s0, false, false);
      const __bf16* kr1 = &kst[buf][(16 + n16) * HDQ + 16 * h];
      v16bf b2 = *(const v16bf*)kr1;
      v16bf b3 = *(const v16bf*)(kr1 + 32);
      s1 = __builtin_amdgcn_wmma_f32_16x16x32_bf16(false, qa0, false, b2, (short)0, s1, false, false);
      s1 = __builtin_amdgcn_wmma_f32_16x16x32_bf16(false, qa1, false, b3, (short)0, s1, false, false);
    }

    // ---- online softmax over the 32 new columns; C layout row = r + 8h, col = n16
#pragma unroll
    for (int r = 0; r < 8; ++r) {
      float v0 = s0[r] * scale;
      float v1 = s1[r] * scale;
      float mx = fmaxf(v0, v1);
#pragma unroll
      for (int d = 8; d >= 1; d >>= 1) mx = fmaxf(mx, __shfl_xor(mx, d, 32));
      const float mnew  = fmaxf(m[r], mx);
      const float alpha = __expf(m[r] - mnew);
      const float p0 = __expf(v0 - mnew);
      const float p1 = __expf(v1 - mnew);
      float rs = p0 + p1;
#pragma unroll
      for (int d = 8; d >= 1; d >>= 1) rs += __shfl_xor(rs, d, 32);
      l[r] = l[r] * alpha + rs;
      m[r] = mnew;
      o[0][r] *= alpha; o[1][r] *= alpha; o[2][r] *= alpha; o[3][r] *= alpha;
      // stage P (16x32) in LDS to convert C-frag -> A-frag layout
      const int row = r + 8 * h;
      plds[wave][row * 32 + n16]      = (__bf16)p0;
      plds[wave][row * 32 + 16 + n16] = (__bf16)p1;
    }

    // ---- O += P(16x32) @ V(32x64)
    const __bf16* pr = &plds[wave][n16 * 32];
    v16bf pa = frag_a_2x8(pr + 16 * h, pr + 16 * h + 8);
#pragma unroll
    for (int j = 0; j < 4; ++j) {
      o[j] = __builtin_amdgcn_wmma_f32_16x16x32_bf16(
          false, pa, false, vf[j], (short)0, o[j], false, false);
    }
    buf ^= 1;
  }

  // ---- normalize and write back (bf16, head-major columns so O-proj A-frags are contiguous)
  float inv[8];
#pragma unroll
  for (int r = 0; r < 8; ++r) inv[r] = 1.0f / l[r];
#pragma unroll
  for (int j = 0; j < 4; ++j) {
    const int d = j * 16 + n16;
#pragma unroll
    for (int r = 0; r < 8; ++r) {
      const int s = sT * 16 + r + 8 * h;
      Ob[(size_t)(b * SQ + s) * EQ + head * HDQ + d] = (__bf16)(o[j][r] * inv[r]);
    }
  }
}

// ---------- host launch ----------

extern "C" void kernel_launch(void* const* d_in, const int* in_sizes, int n_in,
                              void* d_out, int out_size, void* d_ws, size_t ws_size,
                              hipStream_t stream) {
  const float* x  = (const float*)d_in[0];
  const float* Wq = (const float*)d_in[1];
  const float* bq = (const float*)d_in[2];
  const float* Wk = (const float*)d_in[3];
  const float* bk = (const float*)d_in[4];
  const float* Wv = (const float*)d_in[5];
  const float* bv = (const float*)d_in[6];
  const float* Wo = (const float*)d_in[7];
  const float* bo = (const float*)d_in[8];
  float* out = (float*)d_out;

  char* ws = (char*)d_ws;
  size_t off = 0;
  auto alloc = [&](size_t bytes) -> char* {
    char* p = ws + off;
    off += (bytes + 255) & ~(size_t)255;
    return p;
  };
  __bf16* xb  = (__bf16*)alloc((size_t)MTOT * EQ  * 2);   // x in bf16
  __bf16* WqT = (__bf16*)alloc((size_t)EQ   * EQ  * 2);   // [N][K]
  __bf16* WkT = (__bf16*)alloc((size_t)KVQ  * EQ  * 2);
  __bf16* WvT = (__bf16*)alloc((size_t)KVQ  * EQ  * 2);
  __bf16* WoT = (__bf16*)alloc((size_t)EQ   * EQ  * 2);
  __bf16* qb  = (__bf16*)alloc((size_t)MTOT * EQ  * 2);   // Q  [B*S, E]
  __bf16* kb  = (__bf16*)alloc((size_t)MTOT * KVQ * 2);   // K  [B*S, KV]
  __bf16* vt  = (__bf16*)alloc((size_t)MTOT * KVQ * 2);   // V^T [KV, B*S]
  __bf16* ob  = (__bf16*)alloc((size_t)MTOT * EQ  * 2);   // attn out [B*S, E]
  (void)ws_size; (void)in_sizes; (void)n_in; (void)out_size;

  const int nX = MTOT * EQ;
  cvt_bf16_kernel<<<(nX + 255) / 256, 256, 0, stream>>>(x, xb, nX);
  cvt_wT_kernel<<<(EQ * EQ  + 255) / 256, 256, 0, stream>>>(Wq, WqT, EQ, EQ);
  cvt_wT_kernel<<<(EQ * KVQ + 255) / 256, 256, 0, stream>>>(Wk, WkT, EQ, KVQ);
  cvt_wT_kernel<<<(EQ * KVQ + 255) / 256, 256, 0, stream>>>(Wv, WvT, EQ, KVQ);
  cvt_wT_kernel<<<(EQ * EQ  + 255) / 256, 256, 0, stream>>>(Wo, WoT, EQ, EQ);

  auto blocksFor = [](int M, int N) {
    int waves = (M / 16) * (N / 64);
    return (waves + 7) / 8;       // 8 waves per 256-thread block
  };

  // projections: Q, K (row-major bf16), V (transposed bf16)
  gemm_bf16_wmma<<<blocksFor(MTOT, EQ),  256, 0, stream>>>(xb, WqT, bq, qb, MTOT, EQ,  EQ, EQ,  0);
  gemm_bf16_wmma<<<blocksFor(MTOT, KVQ), 256, 0, stream>>>(xb, WkT, bk, kb, MTOT, KVQ, EQ, KVQ, 0);
  gemm_bf16_wmma<<<blocksFor(MTOT, KVQ), 256, 0, stream>>>(xb, WvT, bv, vt, MTOT, KVQ, EQ, KVQ, 1);

  // attention: B*G*GH*(S/16) = 4096 waves, AW per block sharing one K stream
  const int attnTiles = BQ * GQ * GHQ * (SQ / 16);
  gqa_attn_wmma<<<attnTiles / AW, AW * 32, 0, stream>>>(qb, kb, vt, ob);

  // output projection -> f32
  gemm_bf16_wmma<<<blocksFor(MTOT, EQ), 256, 0, stream>>>(ob, WoT, bo, out, MTOT, EQ, EQ, EQ, 2);
}
